// GNN_20839181320257
// MI455X (gfx1250) — matrix operand order repaired and checked by reading, not compiled, not run on hardware
//
#include <hip/hip_runtime.h>
#include <hip/hip_bf16.h>

// ---------------------------------------------------------------------------
// GNN MetaLayer x2 on gfx1250: edge/node MLPs as v_wmma_f32_16x16x32_f16
// ---------------------------------------------------------------------------

#define GN 131072      // nodes
#define GE 2097152     // edges
#define GB 4           // graphs
#define NW 8           // waves (16-row tiles) per block

typedef __attribute__((ext_vector_type(8)))  _Float16 v8h;
typedef __attribute__((ext_vector_type(16))) _Float16 v16h;
typedef __attribute__((ext_vector_type(8)))  float    v8f;

__device__ __forceinline__ v16h join8(v8h lo, v8h hi) {
  v16h r;
#pragma unroll
  for (int i = 0; i < 8; ++i) { r[i] = lo[i]; r[i + 8] = hi[i]; }
  return r;
}

// A-fragment (16xK row-major in LDS, row pointer given). ISA 16-bit A 16x32:
// lanes 0-15: halves 0..7 -> K=0..7, halves 8..15 -> K=16..23; lanes 16-31: +8.
__device__ __forceinline__ v16h load_a(const _Float16* rowp, int s, int lane) {
  const int k0 = s * 32 + ((lane & 16) ? 8 : 0);
  v8h lo = *(const v8h*)(rowp + k0);
  v8h hi = *(const v8h*)(rowp + k0 + 16);
  return join8(lo, hi);
}

// B-fragment from transposed weights W^T[n][k] in LDS (row = output column n).
// ISA B 32x16: lanes 0-15: halves 0..7 -> K=0..7, halves 8..15 -> K=16..23; lanes 16-31: +8.
__device__ __forceinline__ v16h load_b(const _Float16* colp, int s, int lane) {
  const int k0 = s * 32 + ((lane & 16) ? 8 : 0);
  v8h lo = *(const v8h*)(colp + k0);
  v8h hi = *(const v8h*)(colp + k0 + 16);
  return join8(lo, hi);
}

__device__ __forceinline__ v8f wmma_f16(v16h a, v16h b, v8f c) {
  return __builtin_amdgcn_wmma_f32_16x16x32_f16(false, a, false, b,
                                                (short)0, c, false, false);
}

__device__ __forceinline__ float lrelu(float z) { return z >= 0.f ? z : 0.01f * z; }

// ---------------------------------------------------------------------------
__global__ void k_zero(float* __restrict__ p, int n) {
  int i = blockIdx.x * blockDim.x + threadIdx.x;
  if (i < n) p[i] = 0.f;
}

__global__ void k_count(const int* __restrict__ col, const int* __restrict__ batch,
                        float* __restrict__ deg, float* __restrict__ gcnt) {
  int i = blockIdx.x * blockDim.x + threadIdx.x;
  if (i < GE) atomicAdd(&deg[col[i]], 1.f);
  if (i < GN) atomicAdd(&gcnt[batch[i]], 1.f);
}

// ---------------------------------------------------------------------------
// Fused edge model + node mlp_1 + scatter-mean numerator.
// One wave handles 16 edges.
// ---------------------------------------------------------------------------
__global__ __launch_bounds__(256) void k_edge(
    const float* __restrict__ x, const float* __restrict__ ein_attr,
    const int* __restrict__ row, const int* __restrict__ col,
    const int* __restrict__ batch, const float* __restrict__ u,
    const float* __restrict__ ew1, const float* __restrict__ eb1,
    const float* __restrict__ ew2, const float* __restrict__ eb2,
    const float* __restrict__ nw1, const float* __restrict__ nb1,
    const float* __restrict__ nw2, const float* __restrict__ nb2,
    float* __restrict__ eout, float* __restrict__ agg) {
  __shared__ alignas(16) _Float16 s_w1t[64][96];   // ew1^T, K padded 84->96
  __shared__ alignas(16) _Float16 s_w2t[16][64];   // ew2^T, N padded 6->16
  __shared__ alignas(16) _Float16 s_n1w1t[32][32]; // n1w1^T, K padded 13->32
  __shared__ alignas(16) _Float16 s_n1w2t[32][32]; // n1w2^T
  __shared__ alignas(16) _Float16 s_ein[NW][16][96];

  const int tid = threadIdx.x;
  for (int i = tid; i < 64 * 96; i += 256) { int n = i / 96, k = i % 96; s_w1t[n][k] = (_Float16)(k < 84 ? ew1[k * 64 + n] : 0.f); }
  for (int i = tid; i < 16 * 64; i += 256) { int n = i / 64, k = i % 64; s_w2t[n][k] = (_Float16)(n < 6 ? ew2[k * 6 + n] : 0.f); }
  for (int i = tid; i < 32 * 32; i += 256) { int n = i / 32, k = i % 32; s_n1w1t[n][k] = (_Float16)(k < 13 ? nw1[k * 32 + n] : 0.f); }
  for (int i = tid; i < 32 * 32; i += 256) { int n = i / 32, k = i % 32; s_n1w2t[n][k] = (_Float16)nw2[k * 32 + n]; }

  const int w = tid >> 5, lane = tid & 31;
  const int ebase = (blockIdx.x * NW + w) * 16;
  _Float16(*ein)[96] = s_ein[w];

  { // stage e_in = [x[row](7) | x[col](7) | edge_attr(6) | u[batch[row]](64) | pad]
    const int m = lane & 15, e = ebase + m;
    if (lane < 16) {
      const int r0 = row[e], c0 = col[e];
#pragma unroll
      for (int j = 0; j < 7; ++j) ein[m][j] = (_Float16)x[r0 * 7 + j];
#pragma unroll
      for (int j = 0; j < 7; ++j) ein[m][7 + j] = (_Float16)x[c0 * 7 + j];
#pragma unroll
      for (int j = 0; j < 6; ++j) ein[m][14 + j] = (_Float16)ein_attr[(size_t)e * 6 + j];
#pragma unroll
      for (int j = 84; j < 96; ++j) ein[m][j] = (_Float16)0.f;
    } else {
      const int g = batch[row[e]];
#pragma unroll
      for (int j = 0; j < 64; ++j) ein[m][20 + j] = (_Float16)u[g * 64 + j];
    }
  }
  __syncthreads();

  const int mrow = lane & 15;
  const int mhi = (lane & 16) ? 8 : 0;

  // Edge layer 1: [16x96] x [96x64]
  v8f acc[4] = {};
#pragma unroll
  for (int s = 0; s < 3; ++s) {
    v16h a = load_a(&ein[mrow][0], s, lane);
#pragma unroll
    for (int t = 0; t < 4; ++t) {
      v16h b = load_b(&s_w1t[t * 16 + mrow][0], s, lane);
      acc[t] = wmma_f16(a, b, acc[t]);
    }
  }
  __syncthreads();
#pragma unroll
  for (int t = 0; t < 4; ++t) { // bias + leaky -> h1 at cols 32..95
    const int n = t * 16 + mrow;
    const float bb = eb1[n];
#pragma unroll
    for (int r = 0; r < 8; ++r) ein[r + mhi][32 + n] = (_Float16)lrelu(acc[t][r] + bb);
  }
  __syncthreads();

  // Edge layer 2: [16x64] x [64x16(6)]
  v8f c2 = {};
#pragma unroll
  for (int s = 0; s < 2; ++s) {
    v16h a = load_a(&ein[mrow][32], s, lane);
    v16h b = load_b(&s_w2t[mrow][0], s, lane);
    c2 = wmma_f16(a, b, c2);
  }
  __syncthreads();

  // Build nh = [x[row](7) | new edge_attr(6) | pad..31] in cols 0..31, emit edge_attr
  if (lane < 16) {
#pragma unroll
    for (int j = 13; j < 32; ++j) ein[lane][j] = (_Float16)0.f;
  }
  if (mrow < 6) {
    const float bb = eb2[mrow];
#pragma unroll
    for (int r = 0; r < 8; ++r) {
      const int m = r + mhi;
      const float v = c2[r] + bb;
      eout[(size_t)(ebase + m) * 6 + mrow] = v;
      ein[m][7 + mrow] = (_Float16)v;
    }
  }
  __syncthreads();

  // Node mlp_1 layer 1: [16x32(13)] x [32x32]
  v8f a1[2] = {};
  {
    v16h a = load_a(&ein[mrow][0], 0, lane);
#pragma unroll
    for (int t = 0; t < 2; ++t) {
      v16h b = load_b(&s_n1w1t[t * 16 + mrow][0], 0, lane);
      a1[t] = wmma_f16(a, b, a1[t]);
    }
  }
  __syncthreads();
#pragma unroll
  for (int t = 0; t < 2; ++t) {
    const int n = t * 16 + mrow;
    const float bb = nb1[n];
#pragma unroll
    for (int r = 0; r < 8; ++r) ein[r + mhi][32 + n] = (_Float16)lrelu(a1[t][r] + bb);
  }
  __syncthreads();

  // Node mlp_1 layer 2: [16x32] x [32x32]
  v8f a2[2] = {};
  {
    v16h a = load_a(&ein[mrow][32], 0, lane);
#pragma unroll
    for (int t = 0; t < 2; ++t) {
      v16h b = load_b(&s_n1w2t[t * 16 + mrow][0], 0, lane);
      a2[t] = wmma_f16(a, b, a2[t]);
    }
  }

  // Scatter-add nh into agg[col] (segment-sum numerator)
  int cols8[8];
#pragma unroll
  for (int r = 0; r < 8; ++r) cols8[r] = col[ebase + r + mhi];
#pragma unroll
  for (int t = 0; t < 2; ++t) {
    const int n = t * 16 + mrow;
    const float bb = nb2[n];
#pragma unroll
    for (int r = 0; r < 8; ++r)
      atomicAdd(&agg[(size_t)cols8[r] * 32 + n], a2[t][r] + bb);
  }
}

// ---------------------------------------------------------------------------
// Node mlp_2 + per-graph mean numerator. One wave = 16 nodes.
// ---------------------------------------------------------------------------
__global__ __launch_bounds__(256) void k_node(
    const float* __restrict__ x, const float* __restrict__ agg,
    const float* __restrict__ deg, const int* __restrict__ batch,
    const float* __restrict__ u,
    const float* __restrict__ w1, const float* __restrict__ b1,
    const float* __restrict__ w2, const float* __restrict__ b2,
    float* __restrict__ xout, float* __restrict__ gm) {
  __shared__ alignas(16) _Float16 s_w1t[32][128]; // n2w1^T, K padded 103->128
  __shared__ alignas(16) _Float16 s_w2t[16][32];  // n2w2^T, N padded 7->16
  __shared__ alignas(16) _Float16 s_xh[NW][16][128];
  __shared__ float s_gm[32];

  const int tid = threadIdx.x;
  for (int i = tid; i < 32 * 128; i += 256) { int n = i / 128, k = i % 128; s_w1t[n][k] = (_Float16)(k < 103 ? w1[k * 32 + n] : 0.f); }
  for (int i = tid; i < 16 * 32; i += 256) { int n = i / 32, k = i % 32; s_w2t[n][k] = (_Float16)(n < 7 ? w2[k * 7 + n] : 0.f); }
  if (tid < 32) s_gm[tid] = 0.f;

  const int w = tid >> 5, lane = tid & 31;
  const int vbase = (blockIdx.x * NW + w) * 16;
  _Float16(*xh)[128] = s_xh[w];

  { // stage xh = [x(7) | agg/deg(32) | u[batch](64) | pad]
    const int m = lane & 15, v = vbase + m;
    if (lane < 16) {
#pragma unroll
      for (int j = 0; j < 7; ++j) xh[m][j] = (_Float16)x[v * 7 + j];
      float d = deg[v]; d = d > 1.f ? d : 1.f;
      const float inv = 1.f / d;
#pragma unroll
      for (int j = 0; j < 32; ++j) xh[m][7 + j] = (_Float16)(agg[(size_t)v * 32 + j] * inv);
#pragma unroll
      for (int j = 103; j < 128; ++j) xh[m][j] = (_Float16)0.f;
    } else {
      const int g = batch[v];
#pragma unroll
      for (int j = 0; j < 64; ++j) xh[m][39 + j] = (_Float16)u[g * 64 + j];
    }
  }
  __syncthreads();

  const int mrow = lane & 15;
  const int mhi = (lane & 16) ? 8 : 0;

  v8f acc[2] = {};
#pragma unroll
  for (int s = 0; s < 4; ++s) { // [16x128(103)] x [128x32]
    v16h a = load_a(&xh[mrow][0], s, lane);
#pragma unroll
    for (int t = 0; t < 2; ++t) {
      v16h b = load_b(&s_w1t[t * 16 + mrow][0], s, lane);
      acc[t] = wmma_f16(a, b, acc[t]);
    }
  }
  __syncthreads();
#pragma unroll
  for (int t = 0; t < 2; ++t) {
    const int n = t * 16 + mrow;
    const float bb = b1[n];
#pragma unroll
    for (int r = 0; r < 8; ++r) xh[r + mhi][32 + n] = (_Float16)lrelu(acc[t][r] + bb);
  }
  __syncthreads();

  v8f c2 = {};
  { // [16x32] x [32x16(7)]
    v16h a = load_a(&xh[mrow][32], 0, lane);
    v16h b = load_b(&s_w2t[mrow][0], 0, lane);
    c2 = wmma_f16(a, b, c2);
  }
  if (mrow < 7) {
    const float bb = b2[mrow];
#pragma unroll
    for (int r = 0; r < 8; ++r) {
      const int v = vbase + r + mhi;
      const float vv = c2[r] + bb;
      xout[(size_t)v * 7 + mrow] = vv;
      atomicAdd(&s_gm[batch[v] * 7 + mrow], vv);
    }
  }
  __syncthreads();
  if (tid < 28) atomicAdd(&gm[tid], s_gm[tid]);
}

// ---------------------------------------------------------------------------
// Global model: u' = leaky([u | mean_per_graph(x)] @ g_w1 + b1) @ g_w2 + b2
// B=4 rows -> exact scalar f32, one block.
// ---------------------------------------------------------------------------
__global__ __launch_bounds__(256) void k_global(
    const float* __restrict__ u, const float* __restrict__ gm,
    const float* __restrict__ gcnt,
    const float* __restrict__ w1, const float* __restrict__ b1,
    const float* __restrict__ w2, const float* __restrict__ b2,
    float* __restrict__ uout) {
  __shared__ float cat[GB][71];
  __shared__ float h[GB][64];
  const int tid = threadIdx.x;
  for (int i = tid; i < GB * 71; i += 256) {
    const int g = i / 71, k = i % 71;
    cat[g][k] = k < 64 ? u[g * 64 + k] : gm[g * 7 + (k - 64)] / fmaxf(gcnt[g], 1.f);
  }
  __syncthreads();
  const int g = tid >> 6, o = tid & 63;
  float a = b1[o];
  for (int k = 0; k < 71; ++k) a += cat[g][k] * w1[k * 64 + o];
  h[g][o] = lrelu(a);
  __syncthreads();
  float a2 = b2[o];
  for (int k = 0; k < 64; ++k) a2 += h[g][k] * w2[k * 64 + o];
  uout[g * 64 + o] = a2;
}

// ---------------------------------------------------------------------------
// Readout: u2 = leaky(u@m3+b); train-mode BatchNorm over B rows; @m4+b -> [4,15]
// ---------------------------------------------------------------------------
__global__ __launch_bounds__(256) void k_readout(
    const float* __restrict__ u,
    const float* __restrict__ m3w, const float* __restrict__ m3b,
    const float* __restrict__ bng, const float* __restrict__ bnb,
    const float* __restrict__ m4w, const float* __restrict__ m4b,
    float* __restrict__ out) {
  __shared__ float s_u2[GB][64];
  const int tid = threadIdx.x;
  const int g = tid >> 6, o = tid & 63;
  float a = m3b[o];
  for (int k = 0; k < 64; ++k) a += u[g * 64 + k] * m3w[k * 64 + o];
  s_u2[g][o] = lrelu(a);
  __syncthreads();
  if (tid < 64) {
    float mu = 0.f;
    for (int b = 0; b < GB; ++b) mu += s_u2[b][tid];
    mu *= (1.f / GB);
    float var = 0.f;
    for (int b = 0; b < GB; ++b) { const float d = s_u2[b][tid] - mu; var += d * d; }
    var *= (1.f / GB);
    const float inv = rsqrtf(var + 1e-5f);
    for (int b = 0; b < GB; ++b)
      s_u2[b][tid] = (s_u2[b][tid] - mu) * inv * bng[tid] + bnb[tid];
  }
  __syncthreads();
  if (tid < GB * 15) {
    const int b = tid / 15, k = tid % 15;
    float acc = m4b[k];
    for (int j = 0; j < 64; ++j) acc += s_u2[b][j] * m4w[j * 15 + k];
    out[b * 15 + k] = acc;
  }
}

// ---------------------------------------------------------------------------
extern "C" void kernel_launch(void* const* d_in, const int* in_sizes, int n_in,
                              void* d_out, int out_size, void* d_ws, size_t ws_size,
                              hipStream_t stream) {
  (void)in_sizes; (void)n_in; (void)out_size; (void)ws_size;

  const float* x_in   = (const float*)d_in[0];
  const float* ea_in  = (const float*)d_in[1];
  const int*   eidx   = (const int*)d_in[2];
  const int*   batch  = (const int*)d_in[3];
  const float* ew1 = (const float*)d_in[4],  *eb1 = (const float*)d_in[5];
  const float* ew2 = (const float*)d_in[6],  *eb2 = (const float*)d_in[7];
  const float* n1w1 = (const float*)d_in[8], *n1b1 = (const float*)d_in[9];
  const float* n1w2 = (const float*)d_in[10],*n1b2 = (const float*)d_in[11];
  const float* n2w1 = (const float*)d_in[12],*n2b1 = (const float*)d_in[13];
  const float* n2w2 = (const float*)d_in[14],*n2b2 = (const float*)d_in[15];
  const float* gw1 = (const float*)d_in[16], *gb1 = (const float*)d_in[17];
  const float* gw2 = (const float*)d_in[18], *gb2 = (const float*)d_in[19];
  const float* m3w = (const float*)d_in[20], *m3b = (const float*)d_in[21];
  const float* bng = (const float*)d_in[22], *bnb = (const float*)d_in[23];
  const float* m4w = (const float*)d_in[24], *m4b = (const float*)d_in[25];

  const int* row = eidx;
  const int* col = eidx + GE;

  float* ws = (float*)d_ws;
  size_t off = 0;
  float* e_ws = ws + off; off += (size_t)GE * 6;  // 50.3 MB: recurrent edge_attr
  float* deg  = ws + off; off += GN;              // in-degree (constant)
  float* agg  = ws + off; off += (size_t)GN * 32; // scatter-mean numerator
  float* xb0  = ws + off; off += (size_t)GN * 7;
  float* xb1  = ws + off; off += (size_t)GN * 7;
  float* u0   = ws + off; off += 256;
  float* u1   = ws + off; off += 256;
  float* gm   = ws + off; off += 32;
  float* gcnt = ws + off; off += 8;

  // one-time (per call) setup: degrees, graph counts, u = 0
  k_zero<<<(GN + 255) / 256, 256, 0, stream>>>(deg, GN);
  k_zero<<<1, 64, 0, stream>>>(gcnt, 8);
  k_zero<<<1, 256, 0, stream>>>(u0, 256);
  k_count<<<(GE + 255) / 256, 256, 0, stream>>>(col, batch, deg, gcnt);

  const float* xc = x_in;  float* xn = xb0;
  const float* uc = u0;    float* un = u1;
  const float* ea = ea_in;

  for (int r = 0; r < 2; ++r) {
    k_zero<<<(GN * 32 + 255) / 256, 256, 0, stream>>>(agg, GN * 32);
    k_zero<<<1, 32, 0, stream>>>(gm, 32);
    k_edge<<<GE / (16 * NW), 256, 0, stream>>>(
        xc, ea, row, col, batch, uc,
        ew1, eb1, ew2, eb2, n1w1, n1b1, n1w2, n1b2, e_ws, agg);
    k_node<<<GN / (16 * NW), 256, 0, stream>>>(
        xc, agg, deg, batch, uc, n2w1, n2b1, n2w2, n2b2, xn, gm);
    k_global<<<1, 256, 0, stream>>>(uc, gm, gcnt, gw1, gb1, gw2, gb2, un);
    ea = e_ws;
    xc = xn; xn = xb1;
    float* t = (float*)uc; uc = un; un = t;
  }
  k_readout<<<1, 256, 0, stream>>>(uc, m3w, m3b, bng, bnb, m4w, m4b, (float*)d_out);
}
